// SPP_VGG16_67396626808983
// MI455X (gfx1250) — matrix operand order repaired and checked by reading, not compiled, not run on hardware
//
#include <hip/hip_runtime.h>
#include <hip/hip_bf16.h>
#include <math.h>

// ---------------------------------------------------------------------------
// Types for CDNA5 WMMA (wave32): A/B = 16 bf16 per lane, C/D = 8 f32 per lane
// ---------------------------------------------------------------------------
typedef __bf16 bf16_t;
typedef __attribute__((ext_vector_type(16))) bf16_t v16bf;
typedef __attribute__((ext_vector_type(8)))  bf16_t v8bf;
typedef __attribute__((ext_vector_type(8)))  float  v8f;

#define BN_EPS 1e-5f

// ---------------------------------------------------------------------------
// Input image: NCHW f32 -> NHWC bf16 (channels-last for contiguous K-loads)
// ---------------------------------------------------------------------------
__global__ void k_nchw_to_nhwc_bf16(bf16_t* __restrict__ dst, const float* __restrict__ src,
                                    int C, int HW, int n) {
  int i = blockIdx.x * blockDim.x + threadIdx.x;
  if (i >= n) return;
  int c = i % C;
  int p = i / C;              // flat (b,y,x)
  int b = p / HW;
  int rem = p - b * HW;
  dst[i] = (bf16_t)src[((size_t)b * C + c) * HW + rem];
}

// ---------------------------------------------------------------------------
// Fold BN into conv weights and pre-swizzle into the WMMA A-fragment layout.
// K ordering: K = pos*Cin + cin  (pos = ky*3+kx, cin contiguous -> NHWC friendly)
// Packed layout: [coutBlock(16 rows)][kBlock(32 K)][lane(32)][elem(16 bf16)],
// so each lane's fragment is 32 contiguous bytes.
// A frag as packed here (contiguous k, B loads match): lane<16 -> M=lane, K=e(0..15); lane>=16 -> K=16+e.
// ---------------------------------------------------------------------------
__global__ void k_pack_weights(bf16_t* __restrict__ packed, float* __restrict__ biasOut,
                               const float* __restrict__ w, const float* __restrict__ b,
                               const float* __restrict__ g, const float* __restrict__ be,
                               const float* __restrict__ mean, const float* __restrict__ var,
                               int Cin, int Cout, int Kreal, int KB, int P) {
  int tid = blockIdx.x * blockDim.x + threadIdx.x;
  if (tid >= P) return;
  int e    = tid & 15;
  int lane = (tid >> 4) & 31;
  int t2   = tid >> 9;
  int kb   = t2 % KB;
  int cb   = t2 / KB;
  int m    = lane & 15;
  int half = lane >> 4;
  int k32  = e + 16 * half;
  int K    = kb * 32 + k32;
  int cout = cb * 16 + m;
  float scale = g[cout] * rsqrtf(var[cout] + BN_EPS);
  float val = 0.0f;
  if (K < Kreal) {
    int pos = K / Cin;
    int cin = K - pos * Cin;
    val = w[((size_t)cout * Cin + cin) * 9 + pos] * scale;
  }
  packed[tid] = (bf16_t)val;
  if (tid < Cout) {
    float sc = g[tid] * rsqrtf(var[tid] + BN_EPS);
    biasOut[tid] = (b[tid] - mean[tid]) * sc + be[tid];
  }
}

// ---------------------------------------------------------------------------
// Pack fc_w [200 x 5376] f32 into A-fragment layout, rows padded to 208.
// ---------------------------------------------------------------------------
__global__ void k_pack_fc(bf16_t* __restrict__ packed, const float* __restrict__ w,
                          int Mrows, int K, int KB, int P) {
  int tid = blockIdx.x * blockDim.x + threadIdx.x;
  if (tid >= P) return;
  int e    = tid & 15;
  int lane = (tid >> 4) & 31;
  int t2   = tid >> 9;
  int kb   = t2 % KB;
  int cb   = t2 / KB;
  int m    = lane & 15;
  int half = lane >> 4;
  int k32  = e + 16 * half;
  int Kidx = kb * 32 + k32;
  int row  = cb * 16 + m;
  float val = (row < Mrows) ? w[(size_t)row * K + Kidx] : 0.0f;
  packed[tid] = (bf16_t)val;
}

// ---------------------------------------------------------------------------
// Fast conv3x3 (pad 1) + bias + ReLU, NHWC, Cin % 32 == 0.
// One wave: 64 couts x 32 pixels (8 accumulators). Per 32-K step:
//   2 contiguous B loads (32B each) + 4 contiguous A loads, 8 WMMAs.
// No divergent exits: grid sized exactly; EXEC all-1s at every WMMA.
// ---------------------------------------------------------------------------
__global__ __launch_bounds__(256) void k_conv_wmma_nhwc(
    bf16_t* __restrict__ out, const bf16_t* __restrict__ in,
    const bf16_t* __restrict__ packedW, const float* __restrict__ bias,
    int Cin, int Cout, int H, int W, int Ntiles) {
  int wave = blockIdx.x * 8 + (threadIdx.x >> 5);
  int lane = threadIdx.x & 31;
  int cb4  = wave / Ntiles;            // 64-channel output block
  int tile = wave - cb4 * Ntiles;      // 32-pixel tile
  int half = lane >> 4;
  int nloc = lane & 15;
  int HW   = H * W;

  int ng0 = tile * 32 + nloc;          // pixel for sub-tile 0
  int ng1 = ng0 + 16;                  // pixel for sub-tile 1
  int b0 = ng0 / HW, r0 = ng0 - b0 * HW, y0 = r0 / W, x0 = r0 - y0 * W;
  int b1 = ng1 / HW, r1 = ng1 - b1 * HW, y1 = r1 / W, x1 = r1 - y1 * W;

  int CB = Cin >> 5;                   // 32-channel K sub-blocks per position
  int KB = 9 * CB;

  v8f acc[8] = {};
  int kb = 0;
  for (int pos = 0; pos < 9; ++pos) {
    int dy = pos / 3 - 1, dx = pos - (pos / 3) * 3 - 1;
    int iy0 = y0 + dy, ix0 = x0 + dx;
    int iy1 = y1 + dy, ix1 = x1 + dx;
    bool v0 = (unsigned)iy0 < (unsigned)H && (unsigned)ix0 < (unsigned)W;
    bool v1 = (unsigned)iy1 < (unsigned)H && (unsigned)ix1 < (unsigned)W;
    const bf16_t* s0 = in + (((size_t)b0 * H + iy0) * W + ix0) * Cin + half * 16;
    const bf16_t* s1 = in + (((size_t)b1 * H + iy1) * W + ix1) * Cin + half * 16;
    for (int cbK = 0; cbK < CB; ++cbK, ++kb) {
      v16bf bf0 = {}, bf1 = {};
      if (v0) bf0 = *(const v16bf*)(s0 + cbK * 32);
      if (v1) bf1 = *(const v16bf*)(s1 + cbK * 32);
      if (kb + 1 < KB)                 // warm next K-step's weights
        __builtin_prefetch(packedW + (((size_t)(cb4 * 4) * KB + (kb + 1)) * 32 + lane) * 16, 0, 3);
#pragma unroll
      for (int j = 0; j < 4; ++j) {
        v16bf a = *(const v16bf*)(packedW + (((size_t)(cb4 * 4 + j) * KB + kb) * 32 + lane) * 16);
        acc[j]     = __builtin_amdgcn_wmma_f32_16x16x32_bf16(false, a, false, bf0, (short)0, acc[j],     false, false);
        acc[j + 4] = __builtin_amdgcn_wmma_f32_16x16x32_bf16(false, a, false, bf1, (short)0, acc[j + 4], false, false);
      }
    }
  }

  // epilogue: vector bias, ReLU, pack 8 bf16, one b128 store per accumulator
  bf16_t* o0 = out + (size_t)ng0 * Cout + cb4 * 64 + half * 8;
  bf16_t* o1 = out + (size_t)ng1 * Cout + cb4 * 64 + half * 8;
#pragma unroll
  for (int j = 0; j < 4; ++j) {
    v8f bv = *(const v8f*)(bias + cb4 * 64 + j * 16 + half * 8);
    v8bf p0, p1;
#pragma unroll
    for (int r = 0; r < 8; ++r) {
      float a = acc[j][r] + bv[r];
      float b = acc[j + 4][r] + bv[r];
      p0[r] = (bf16_t)(a > 0.0f ? a : 0.0f);
      p1[r] = (bf16_t)(b > 0.0f ? b : 0.0f);
    }
    *(v8bf*)(o0 + j * 16) = p0;
    *(v8bf*)(o1 + j * 16) = p1;
  }
}

// ---------------------------------------------------------------------------
// Gather-path conv (layer 0 only: Cin=3, K=27 padded to 32), NHWC.
// One wave: 64 couts x 16 pixels.
// ---------------------------------------------------------------------------
__global__ __launch_bounds__(256) void k_conv_wmma_gather(
    bf16_t* __restrict__ out, const bf16_t* __restrict__ in,
    const bf16_t* __restrict__ packedW, const float* __restrict__ bias,
    int Cin, int Cout, int H, int W, int Kreal, int KB, int Ntiles) {
  int wave = blockIdx.x * 8 + (threadIdx.x >> 5);
  int lane = threadIdx.x & 31;
  int cb4  = wave / Ntiles;
  int tile = wave - cb4 * Ntiles;
  int half = lane >> 4;
  int nloc = lane & 15;
  int HW   = H * W;
  int ng   = tile * 16 + nloc;
  int bimg = ng / HW;
  int rem  = ng - bimg * HW;
  int y    = rem / W;
  int x    = rem - y * W;

  v8f acc[4] = {};
  for (int kb = 0; kb < KB; ++kb) {
    v16bf bfrag;
    int Kbase = kb * 32 + half * 16;
#pragma unroll
    for (int e = 0; e < 16; ++e) {
      int K = Kbase + e;
      float v = 0.0f;
      if (K < Kreal) {
        int pos = K / Cin;
        int cin = K - pos * Cin;
        int iy = y + pos / 3 - 1;
        int ix = x + (pos - (pos / 3) * 3) - 1;
        if ((unsigned)iy < (unsigned)H && (unsigned)ix < (unsigned)W)
          v = (float)in[(((size_t)bimg * H + iy) * W + ix) * Cin + cin];
      }
      bfrag[e] = (bf16_t)v;
    }
#pragma unroll
    for (int j = 0; j < 4; ++j) {
      v16bf a = *(const v16bf*)(packedW + (((size_t)(cb4 * 4 + j) * KB + kb) * 32 + lane) * 16);
      acc[j] = __builtin_amdgcn_wmma_f32_16x16x32_bf16(false, a, false, bfrag, (short)0, acc[j], false, false);
    }
  }
  bf16_t* op = out + (size_t)ng * Cout + cb4 * 64 + half * 8;
#pragma unroll
  for (int j = 0; j < 4; ++j) {
    v8f bv = *(const v8f*)(bias + cb4 * 64 + j * 16 + half * 8);
    v8bf p;
#pragma unroll
    for (int r = 0; r < 8; ++r) {
      float a = acc[j][r] + bv[r];
      p[r] = (bf16_t)(a > 0.0f ? a : 0.0f);
    }
    *(v8bf*)(op + j * 16) = p;
  }
}

// ---------------------------------------------------------------------------
// 2x2 stride-2 maxpool, NHWC, vectorized over 8 contiguous channels
// ---------------------------------------------------------------------------
__global__ void k_maxpool2_nhwc(bf16_t* __restrict__ out, const bf16_t* __restrict__ in,
                                int C, int Ho, int Wo, int n8) {
  int i = blockIdx.x * blockDim.x + threadIdx.x;
  if (i >= n8) return;
  int G = C >> 3;
  int g = i % G; int t = i / G;
  int xo = t % Wo; t /= Wo;
  int yo = t % Ho; int b = t / Ho;
  int Wi = Wo * 2;
  size_t rowC = (size_t)Wi * C;
  const bf16_t* p = in + (((size_t)b * (Ho * 2) + yo * 2) * Wi + xo * 2) * C + g * 8;
  v8bf a0 = *(const v8bf*)p;
  v8bf a1 = *(const v8bf*)(p + C);
  v8bf a2 = *(const v8bf*)(p + rowC);
  v8bf a3 = *(const v8bf*)(p + rowC + C);
  v8bf o;
#pragma unroll
  for (int r = 0; r < 8; ++r) {
    float m = fmaxf(fmaxf((float)a0[r], (float)a1[r]), fmaxf((float)a2[r], (float)a3[r]));
    o[r] = (bf16_t)m;
  }
  *(v8bf*)(out + (size_t)i * 8) = o;
}

// ---------------------------------------------------------------------------
// SPP over NHWC bf16 feats -> pooled bf16 [48][5376] (max of bf16 is exact).
// pooled[n][lvl_off + c*L*L + i*L + j] matches the reference concat order.
// ---------------------------------------------------------------------------
__global__ void k_spp(bf16_t* __restrict__ pooled, const bf16_t* __restrict__ feats,
                      const float* __restrict__ targets) {
  int tid = blockIdx.x * blockDim.x + threadIdx.x;
  if (tid >= 48 * 256) return;
  int n = tid >> 8;
  int c = tid & 255;
  const float* t = targets + n * 6;
  const int h = 56, w = 56, C = 256;
  int img = (int)t[0];
  int x0  = (int)floorf(t[2] * w);
  int y0  = (int)floorf(t[3] * h);
  int bw  = (int)floorf(t[4] * w); if (bw < 4) bw = 4;
  int bh  = (int)floorf(t[5] * h); if (bh < 4) bh = 4;
  const bf16_t* F = feats + (size_t)img * h * w * C + c;
  bf16_t* outp = pooled + (size_t)n * 5376;
  const int levels[3]  = {1, 2, 4};
  const int lvl_off[3] = {0, 256, 1280};
#pragma unroll
  for (int li = 0; li < 3; ++li) {
    int L = levels[li];
    int qh = bh / L, rh = bh % L;
    int qw = bw / L, rw = bw % L;
    for (int i = 0; i < L; ++i) {
      int sh = y0 + i * qh; int eh = sh + qh + rh;
      if (sh < 0) sh = 0;   if (eh > h) eh = h;
      for (int j = 0; j < L; ++j) {
        int sw = x0 + j * qw; int ew = sw + qw + rw;
        if (sw < 0) sw = 0;   if (ew > w) ew = w;
        float m = -INFINITY;
        for (int yy = sh; yy < eh; ++yy)
          for (int xx = sw; xx < ew; ++xx) {
            float v = (float)F[((size_t)yy * w + xx) * C];
            m = v > m ? v : m;
          }
        outp[lvl_off[li] + c * L * L + i * L + j] = (bf16_t)m;
      }
    }
  }
}

// ---------------------------------------------------------------------------
// FC via WMMA: logits[48][200] = pooled(bf16) @ fc_w^T + fc_b.
// 13 row-blocks (208 padded classes) x 3 box-tiles = 39 waves.
// B frag = contiguous 32B load from the box's pooled row.
// ---------------------------------------------------------------------------
__global__ __launch_bounds__(256) void k_fc_wmma(
    float* __restrict__ logits, const bf16_t* __restrict__ pooledb,
    const bf16_t* __restrict__ packedFC, const float* __restrict__ fc_b) {
  int wave = blockIdx.x * 8 + (threadIdx.x >> 5);
  if (wave >= 39) return;              // wave-uniform: EXEC stays all-1s below
  int lane = threadIdx.x & 31;
  int cb = wave / 3;
  int nt = wave - cb * 3;
  int half = lane >> 4;
  int nloc = lane & 15;
  int box  = nt * 16 + nloc;
  const int K = 5376, KB = 168;
  v8f acc = {};
  const bf16_t* prow = pooledb + (size_t)box * K + half * 16;
  for (int kb = 0; kb < KB; ++kb) {
    v16bf bfrag = *(const v16bf*)(prow + kb * 32);
    v16bf a = *(const v16bf*)(packedFC + ((size_t)cb * KB + kb) * 512 + lane * 16);
    acc = __builtin_amdgcn_wmma_f32_16x16x32_bf16(false, a, false, bfrag, (short)0, acc, false, false);
  }
#pragma unroll
  for (int r = 0; r < 8; ++r) {
    int cl = cb * 16 + half * 8 + r;
    if (cl < 200) logits[(size_t)box * 200 + cl] = acc[r] + fc_b[cl];
  }
}

// ---------------------------------------------------------------------------
// softmax -> probs, then faithful double-softmax NLL: nll = lse(probs) - probs[lab]
// ---------------------------------------------------------------------------
__global__ void k_softmax_nll(float* __restrict__ probs, float* __restrict__ nll,
                              const float* __restrict__ logits,
                              const float* __restrict__ targets) {
  int n = blockIdx.x * blockDim.x + threadIdx.x;
  if (n >= 48) return;
  const float* l = logits + (size_t)n * 200;
  float m = -INFINITY;
  for (int i = 0; i < 200; ++i) m = fmaxf(m, l[i]);
  float s = 0.0f;
  for (int i = 0; i < 200; ++i) s += expf(l[i] - m);
  float inv = 1.0f / s;
  float* pr = probs + (size_t)n * 200;
  for (int i = 0; i < 200; ++i) pr[i] = expf(l[i] - m) * inv;
  float m2 = -INFINITY;
  for (int i = 0; i < 200; ++i) m2 = fmaxf(m2, pr[i]);
  float s2 = 0.0f;
  for (int i = 0; i < 200; ++i) s2 += expf(pr[i] - m2);
  float lse = m2 + logf(s2);
  int lab = (int)targets[n * 6 + 1];
  nll[n] = lse - pr[lab];
}

__global__ void k_loss(float* __restrict__ loss, const float* __restrict__ nll) {
  if (blockIdx.x == 0 && threadIdx.x == 0) {
    float s = 0.0f;
    for (int i = 0; i < 48; ++i) s += nll[i];
    loss[0] = s / 48.0f;  // = (1/B) * sum of per-image means (12 boxes each)
  }
}

// ---------------------------------------------------------------------------
// Host orchestration
// ---------------------------------------------------------------------------
extern "C" void kernel_launch(void* const* d_in, const int* in_sizes, int n_in,
                              void* d_out, int out_size, void* d_ws, size_t ws_size,
                              hipStream_t stream) {
  (void)in_sizes; (void)n_in; (void)out_size; (void)ws_size;

  const float* x       = (const float*)d_in[0];
  const float* targets = (const float*)d_in[1];
  const float* fc_w    = (const float*)d_in[44];
  const float* fc_b    = (const float*)d_in[45];

  static const int CIN[7]  = {3, 64, 64, 128, 128, 256, 256};
  static const int COUT[7] = {64, 64, 128, 128, 256, 256, 256};
  static const int HH[7]   = {224, 224, 112, 112, 56, 56, 56};

  char* ws = (char*)d_ws;
  size_t off = 0;
  auto carve = [&](size_t bytes) -> char* {
    char* p = ws + off;
    off += (bytes + 255) & ~(size_t)255;
    return p;
  };

  const size_t ACT_ELEMS = (size_t)4 * 64 * 224 * 224;
  bf16_t* bufA = (bf16_t*)carve(ACT_ELEMS * sizeof(bf16_t));
  bf16_t* bufB = (bf16_t*)carve(ACT_ELEMS * sizeof(bf16_t));

  bf16_t* packed[7]; float* biasArr[7];
  int KB[7], Kreal[7];
  for (int l = 0; l < 7; ++l) {
    Kreal[l] = CIN[l] * 9;
    KB[l]    = (Kreal[l] + 31) / 32;
    size_t P = (size_t)(COUT[l] / 16) * KB[l] * 512;
    packed[l]  = (bf16_t*)carve(P * sizeof(bf16_t));
    biasArr[l] = (float*)carve(COUT[l] * sizeof(float));
  }
  const int FC_KB = 5376 / 32, FC_CB = 13;                  // 208 padded rows
  int PFC = FC_CB * FC_KB * 512;
  bf16_t* packedFC = (bf16_t*)carve((size_t)PFC * sizeof(bf16_t));
  bf16_t* pooled   = (bf16_t*)carve((size_t)48 * 5376 * sizeof(bf16_t));
  float*  logits   = (float*)carve((size_t)48 * 200 * sizeof(float));
  float*  nll      = (float*)carve(48 * sizeof(float));

  // 1) fold BN + pre-swizzle conv weights; pack fc_w
  for (int l = 0; l < 7; ++l) {
    const float* w    = (const float*)d_in[2 + 6 * l + 0];
    const float* b    = (const float*)d_in[2 + 6 * l + 1];
    const float* g    = (const float*)d_in[2 + 6 * l + 2];
    const float* be   = (const float*)d_in[2 + 6 * l + 3];
    const float* mean = (const float*)d_in[2 + 6 * l + 4];
    const float* var  = (const float*)d_in[2 + 6 * l + 5];
    int P = (COUT[l] / 16) * KB[l] * 512;
    k_pack_weights<<<(P + 255) / 256, 256, 0, stream>>>(
        packed[l], biasArr[l], w, b, g, be, mean, var, CIN[l], COUT[l], Kreal[l], KB[l], P);
  }
  k_pack_fc<<<(PFC + 255) / 256, 256, 0, stream>>>(packedFC, fc_w, 200, 5376, FC_KB, PFC);

  // 2) input f32 NCHW -> bf16 NHWC
  int n0 = 4 * 224 * 224 * 3;
  k_nchw_to_nhwc_bf16<<<(n0 + 255) / 256, 256, 0, stream>>>(bufA, x, 3, 224 * 224, n0);

  // 3) conv stack (ping-pong); maxpool after layers 1 and 3
  bf16_t* cur = bufA; bf16_t* nxt = bufB;
  for (int l = 0; l < 7; ++l) {
    int H = HH[l], W = HH[l];
    int Npix = 4 * H * W;
    if (l == 0) {
      int Ntiles = Npix / 16;
      int waves  = (COUT[l] / 64) * Ntiles;          // multiple of 8
      k_conv_wmma_gather<<<waves / 8, 256, 0, stream>>>(
          nxt, cur, packed[l], biasArr[l], CIN[l], COUT[l], H, W, Kreal[l], KB[l], Ntiles);
    } else {
      int Ntiles = Npix / 32;
      int waves  = (COUT[l] / 64) * Ntiles;          // multiple of 8
      k_conv_wmma_nhwc<<<waves / 8, 256, 0, stream>>>(
          nxt, cur, packed[l], biasArr[l], CIN[l], COUT[l], H, W, Ntiles);
    }
    { bf16_t* t = cur; cur = nxt; nxt = t; }
    if (l == 1 || l == 3) {
      int Ho = H / 2, Wo = W / 2, C = COUT[l];
      int n8 = 4 * Ho * Wo * (C / 8);
      k_maxpool2_nhwc<<<(n8 + 255) / 256, 256, 0, stream>>>(nxt, cur, C, Ho, Wo, n8);
      { bf16_t* t = cur; cur = nxt; nxt = t; }
    }
  }
  bf16_t* feats = cur;   // NHWC [4,56,56,256] bf16

  // 4) SPP -> FC(WMMA) -> softmax/NLL -> loss
  k_spp<<<(48 * 256 + 255) / 256, 256, 0, stream>>>(pooled, feats, targets);
  k_fc_wmma<<<5, 256, 0, stream>>>(logits, pooled, packedFC, fc_b);

  float* outf = (float*)d_out;               // [loss, probs(48*200)]
  k_softmax_nll<<<1, 64, 0, stream>>>(outf + 1, nll, logits, targets);
  k_loss<<<1, 32, 0, stream>>>(outf, nll);
}
